// ResidualCrossCorrelationSelfAttention_88141318848696
// MI455X (gfx1250) — matrix operand order repaired and checked by reading, not compiled, not run on hardware
//
#include <hip/hip_runtime.h>
#include <hip/hip_bf16.h>
#include <math.h>

typedef _Float16 h8  __attribute__((ext_vector_type(8)));
typedef _Float16 v16h __attribute__((ext_vector_type(16)));
typedef float    v8f  __attribute__((ext_vector_type(8)));
typedef int      vi4  __attribute__((vector_size(16)));   // 16-byte int vector

#define B_   4
#define N_   4096
#define D_   384
#define KB_  32          // keys per block iteration
#define QW_  16          // query rows per wave
#define NWAVE_ 8
#define QBLK_ (QW_*NWAVE_)   // 128 query rows per workgroup
#define DCHUNKS_ (D_/32)     // 12
#define DTILES_  (D_/16)     // 24
#define NKB_ (N_/KB_)        // 128 key blocks

__device__ __forceinline__ v8f wmma_f16(v16h a, v16h b, v8f c) {
  return __builtin_amdgcn_wmma_f32_16x16x32_f16(false, a, false, b, (short)0, c, false, false);
}

__device__ __forceinline__ v16h mk16(h8 lo, h8 hp) {
  v16h a;
#pragma unroll
  for (int i = 0; i < 8; ++i) { a[i] = lo[i]; a[i + 8] = hp[i]; }
  return a;
}

// ---- CDNA5 async global->LDS (ASYNCcnt path), 16 bytes per lane ------------
__device__ __forceinline__ void async_ld16(const _Float16* g, _Float16* l) {
#if __has_builtin(__builtin_amdgcn_global_load_async_to_lds_b128)
  __builtin_amdgcn_global_load_async_to_lds_b128(
      (__attribute__((address_space(1))) vi4*)g,
      (__attribute__((address_space(3))) vi4*)l, 0, 0);
#else
  *(h8*)l = *(const h8*)g;   // synchronous fallback
#endif
}

__device__ __forceinline__ void wait_async0() {
#if __has_builtin(__builtin_amdgcn_s_wait_asynccnt)
  __builtin_amdgcn_s_wait_asynccnt(0);
#else
  asm volatile("s_wait_asynccnt 0x0" ::: "memory");
#endif
}

// ---------------------------------------------------------------------------
// Kernel 1: per-row L2 norm; emit f16 normalized Q/K (row major, (B*N, D))
// and f16 V transposed ((B, D, N)) so PV B-fragments are contiguous.
// ---------------------------------------------------------------------------
__global__ __launch_bounds__(256)
void prep_norm_kernel(const float* __restrict__ x,
                      _Float16* __restrict__ qk,
                      _Float16* __restrict__ vt) {
  const int wave = threadIdx.x >> 5;
  const int lane = threadIdx.x & 31;
  const int row  = blockIdx.x * 8 + wave;       // 0 .. B_*N_-1
  if (row >= B_ * N_) return;

  const float* xr = x + (size_t)row * D_;
  float vals[DCHUNKS_];
  float ss = 0.f;
#pragma unroll
  for (int k = 0; k < DCHUNKS_; ++k) {
    float v = xr[lane + k * 32];
    vals[k] = v;
    ss += v * v;
  }
  ss += __shfl_xor(ss, 1,  32);
  ss += __shfl_xor(ss, 2,  32);
  ss += __shfl_xor(ss, 4,  32);
  ss += __shfl_xor(ss, 8,  32);
  ss += __shfl_xor(ss, 16, 32);
  const float inv = 1.0f / fmaxf(sqrtf(ss), 1e-12f);

  const int b = row / N_;
  const int n = row % N_;
  _Float16* qr = qk + (size_t)row * D_;
  _Float16* vb = vt + (size_t)b * D_ * N_;
#pragma unroll
  for (int k = 0; k < DCHUNKS_; ++k) {
    const int d = lane + k * 32;
    qr[d] = (_Float16)(vals[k] * inv);
    vb[(size_t)d * N_ + n] = (_Float16)vals[k];
  }
}

// ---------------------------------------------------------------------------
// Kernel 2: flash attention with double-buffered async K/V^T staging.
// ---------------------------------------------------------------------------
__global__ __launch_bounds__(256)
void flash_attn_kernel(const float* __restrict__ x,
                       const _Float16* __restrict__ qk,
                       const _Float16* __restrict__ vt,
                       float* __restrict__ out) {
  __shared__ _Float16 lds_k[2][KB_ * D_];        // 2 x 24 KB (row = key)
  __shared__ _Float16 lds_vt[2][D_ * KB_];       // 2 x 24 KB (row = d)
  __shared__ _Float16 lds_p[NWAVE_ * QW_ * KB_]; // 8 KB per-wave P staging

  const int tid  = threadIdx.x;
  const int wave = tid >> 5;
  const int lane = tid & 31;
  const int l16  = lane & 15;
  const int hi   = lane >> 4;

  const int b     = blockIdx.x / (N_ / QBLK_);
  const int qb    = blockIdx.x % (N_ / QBLK_);
  const int qrow0 = qb * QBLK_ + wave * QW_;

  const _Float16* kg = qk + (size_t)b * N_ * D_;
  const _Float16* vg = vt + (size_t)b * D_ * N_;

  // ---- load this wave's Q tile as 12 A-fragments (16x32, contiguous k = hi*16 + s)
  v16h qa[DCHUNKS_];
  {
    const _Float16* qbase = qk + ((size_t)(b * N_ + qrow0 + l16)) * D_;
#pragma unroll
    for (int dc = 0; dc < DCHUNKS_; ++dc) {
      const int d0 = dc * 32;
      h8 lo = *(const h8*)(qbase + d0 + hi * 16);       // K = hi*16 .. hi*16+7
      h8 hp = *(const h8*)(qbase + d0 + hi * 16 + 8);   // K = hi*16+8 ..
      qa[dc] = mk16(lo, hp);
    }
  }

  v8f acc[DTILES_];
#pragma unroll
  for (int t = 0; t < DTILES_; ++t) {
#pragma unroll
    for (int r = 0; r < 8; ++r) acc[t][r] = 0.0f;
  }
  float mrun[8], lrun[8];
#pragma unroll
  for (int r = 0; r < 8; ++r) { mrun[r] = -3.0e38f; lrun[r] = 0.0f; }

  const float SCALE = 1.0f / 0.07f;

  // ---- issue async copy of a key block into (lk, lv)
  auto issue_block = [&](int kb0, _Float16* lk, _Float16* lv) {
#pragma unroll
    for (int j = 0; j < 6; ++j) {                 // K block: 1536 h8
      const int idx = tid + j * 256;
      const int row = idx / 48;
      const int col = (idx % 48) * 8;
      async_ld16(kg + (size_t)(kb0 + row) * D_ + col, lk + row * D_ + col);
    }
#pragma unroll
    for (int j = 0; j < 6; ++j) {                 // V^T block: 1536 h8
      const int idx  = tid + j * 256;
      const int d    = idx >> 2;
      const int part = (idx & 3) * 8;
      async_ld16(vg + (size_t)d * N_ + kb0 + part, lv + d * KB_ + part);
    }
  };

  issue_block(0, lds_k[0], lds_vt[0]);
  int cur = 0;

  for (int i = 0; i < NKB_; ++i) {
    wait_async0();       // this wave's copies into buf[cur] done
    __syncthreads();     // all waves' copies done; prev compute on buf[cur^1] done

    if (i + 1 < NKB_)
      issue_block((i + 1) * KB_, lds_k[cur ^ 1], lds_vt[cur ^ 1]);

    const _Float16* lk = lds_k[cur];
    const _Float16* lv = lds_vt[cur];

    // ---- S = Q * K^T : two 16x16 tiles
    v8f s0, s1;
#pragma unroll
    for (int r = 0; r < 8; ++r) { s0[r] = 0.0f; s1[r] = 0.0f; }
#pragma unroll
    for (int dc = 0; dc < DCHUNKS_; ++dc) {
      const int d0 = dc * 32;
      const _Float16* kr0 = lk + l16 * D_ + d0 + hi * 16;
      const _Float16* kr1 = lk + (16 + l16) * D_ + d0 + hi * 16;
      v16h b0 = mk16(*(const h8*)kr0, *(const h8*)(kr0 + 8));
      v16h b1 = mk16(*(const h8*)kr1, *(const h8*)(kr1 + 8));
      s0 = wmma_f16(qa[dc], b0, s0);
      s1 = wmma_f16(qa[dc], b1, s1);
    }

    // ---- online softmax update (row = r + 8*hi, spread over 16-lane half)
    float ef[8];
#pragma unroll
    for (int r = 0; r < 8; ++r) {
      const float a0 = s0[r] * SCALE;
      const float a1 = s1[r] * SCALE;
      float t = fmaxf(a0, a1);
      t = fmaxf(t, __shfl_xor(t, 1, 32));
      t = fmaxf(t, __shfl_xor(t, 2, 32));
      t = fmaxf(t, __shfl_xor(t, 4, 32));
      t = fmaxf(t, __shfl_xor(t, 8, 32));
      const float mn = fmaxf(mrun[r], t);
      ef[r] = __expf(mrun[r] - mn);
      mrun[r] = mn;
      const float p0 = __expf(a0 - mn);
      const float p1 = __expf(a1 - mn);
      float su = p0 + p1;
      su += __shfl_xor(su, 1, 32);
      su += __shfl_xor(su, 2, 32);
      su += __shfl_xor(su, 4, 32);
      su += __shfl_xor(su, 8, 32);
      lrun[r] = lrun[r] * ef[r] + su;
      const int pb = wave * (QW_ * KB_) + (r + 8 * hi) * KB_;
      lds_p[pb + l16]      = (_Float16)p0;
      lds_p[pb + 16 + l16] = (_Float16)p1;
    }

    // ---- rescale accumulators
#pragma unroll
    for (int dt = 0; dt < DTILES_; ++dt) {
#pragma unroll
      for (int r = 0; r < 8; ++r) acc[dt][r] *= ef[r];
    }

    // ---- reload P as an A-fragment (16x32 f16)
    const _Float16* prow = &lds_p[wave * (QW_ * KB_) + l16 * KB_];
    v16h pa = mk16(*(const h8*)(prow + hi * 16),
                   *(const h8*)(prow + hi * 16 + 8));

    // ---- PV: accumulate 24 output d-tiles
#pragma unroll
    for (int dt = 0; dt < DTILES_; ++dt) {
      const _Float16* vr = lv + (dt * 16 + l16) * KB_ + hi * 16;
      v16h bb = mk16(*(const h8*)vr, *(const h8*)(vr + 8));
      acc[dt] = wmma_f16(pa, bb, acc[dt]);
    }

    cur ^= 1;
  }

  // ---- epilogue: out = x + 0.3 * (acc / l)
#pragma unroll
  for (int dt = 0; dt < DTILES_; ++dt) {
#pragma unroll
    for (int r = 0; r < 8; ++r) {
      const int row = qrow0 + r + 8 * hi;
      const int col = dt * 16 + l16;
      const size_t g = ((size_t)(b * N_ + row)) * D_ + col;
      out[g] = x[g] + 0.3f * acc[dt][r] / lrun[r];
    }
  }
}

extern "C" void kernel_launch(void* const* d_in, const int* in_sizes, int n_in,
                              void* d_out, int out_size, void* d_ws, size_t ws_size,
                              hipStream_t stream) {
  (void)in_sizes; (void)n_in; (void)out_size; (void)ws_size;
  const float* x = (const float*)d_in[0];
  float* out = (float*)d_out;

  const size_t halves = (size_t)B_ * N_ * D_;   // 6,291,456 halves each
  _Float16* qk = (_Float16*)d_ws;
  _Float16* vt = (_Float16*)((char*)d_ws + halves * sizeof(_Float16));

  prep_norm_kernel<<<(B_ * N_) / 8, 256, 0, stream>>>(x, qk, vt);

  const int grid = B_ * (N_ / QBLK_);  // 128
  flash_attn_kernel<<<grid, 256, 0, stream>>>(x, qk, vt, out);
}
